// GNNConditioner_56186762166749
// MI455X (gfx1250) — matrix operand order repaired and checked by reading, not compiled, not run on hardware
//
#include <hip/hip_runtime.h>
#include <hip/hip_bf16.h>

// ---------------------------------------------------------------------------
// Problem constants (from the reference)
// ---------------------------------------------------------------------------
static constexpr int Bc   = 8;
static constexpr int NBc  = 4096;
static constexpr int HIDc = 512;
static constexpr int Ec   = 65536;
static constexpr int Nn   = Bc * NBc;        // 32768 nodes

typedef __attribute__((ext_vector_type(16))) __bf16 bf16x16;
typedef __attribute__((ext_vector_type(8)))  float  floatx8;

// ---------------------------------------------------------------------------
// Small helper kernels (normalization / aggregation / dense layer 1 / head)
// ---------------------------------------------------------------------------

__global__ void k_deg_init(float* __restrict__ deg, int n) {
    int t = blockIdx.x * blockDim.x + threadIdx.x;
    if (t < n) deg[t] = 1.0f;                 // self loop
}

__global__ void k_deg_edges(const int* __restrict__ edge, float* __restrict__ deg) {
    int t = blockIdx.x * blockDim.x + threadIdx.x;   // over B*E
    if (t >= Bc * Ec) return;
    int e = t & (Ec - 1);
    int b = t >> 16;                                 // t / E
    int dst = edge[Ec + e] + b * NBc;
    atomicAdd(&deg[dst], 1.0f);
}

__global__ void k_dinv(float* __restrict__ dinv, int n) {
    int t = blockIdx.x * blockDim.x + threadIdx.x;
    if (t >= n) return;
    float d = dinv[t];
    dinv[t] = (d > 0.0f) ? rsqrtf(d) : 0.0f;
}

__global__ void k_ax_init(const float* __restrict__ X, const float* __restrict__ dinv,
                          float* __restrict__ ax, int n) {
    int t = blockIdx.x * blockDim.x + threadIdx.x;   // over N*2
    if (t >= n * 2) return;
    float d = dinv[t >> 1];
    ax[t] = d * d * X[t];
}

__global__ void k_ax_edges(const int* __restrict__ edge, const float* __restrict__ X,
                           const float* __restrict__ dinv, float* __restrict__ ax) {
    int t = blockIdx.x * blockDim.x + threadIdx.x;   // over B*E
    if (t >= Bc * Ec) return;
    int e = t & (Ec - 1);
    int b = t >> 16;
    int src = edge[e] + b * NBc;
    int dst = edge[Ec + e] + b * NBc;
    float nrm = dinv[src] * dinv[dst];
    atomicAdd(&ax[dst * 2 + 0], nrm * X[src * 2 + 0]);
    atomicAdd(&ax[dst * 2 + 1], nrm * X[src * 2 + 1]);
}

__global__ void k_layer1(const float* __restrict__ X, const float* __restrict__ ax,
                         const float* __restrict__ W10, const float* __restrict__ b10,
                         const float* __restrict__ W11, const float* __restrict__ b11,
                         __bf16* __restrict__ H1) {
    int t = blockIdx.x * blockDim.x + threadIdx.x;   // over N*HID
    if (t >= Nn * HIDc) return;
    int n = t >> 9;
    int j = t & (HIDc - 1);
    float x0 = X[n * 2 + 0], x1 = X[n * 2 + 1];
    float a0 = ax[n * 2 + 0], a1 = ax[n * 2 + 1];
    float h = x0 * W10[j] + x1 * W10[HIDc + j] + b10[j]
            + a0 * W11[j] + a1 * W11[HIDc + j] + b11[j];
    H1[t] = (__bf16)(h / (1.0f + __expf(-h)));
}

// transpose-convert weight (K x N f32, row-major) -> (N x K bf16, K-contiguous)
__global__ void k_wt_bf16(const float* __restrict__ W, __bf16* __restrict__ Wt, int K, int N) {
    int t = blockIdx.x * blockDim.x + threadIdx.x;   // over N*K
    if (t >= N * K) return;
    int n = t / K, k = t - n * K;
    Wt[t] = (__bf16)W[(size_t)k * N + n];
}

__global__ void k_agg_init(const __bf16* __restrict__ H1, const float* __restrict__ dinv,
                           float* __restrict__ AH) {
    int t = blockIdx.x * blockDim.x + threadIdx.x;   // over N*HID
    if (t >= Nn * HIDc) return;
    float d = dinv[t >> 9];
    AH[t] = d * d * (float)H1[t];
}

// one thread handles 4 consecutive features of one batched edge
__global__ void k_agg_edges(const int* __restrict__ edge, const __bf16* __restrict__ H1,
                            const float* __restrict__ dinv, float* __restrict__ AH) {
    long long t = (long long)blockIdx.x * blockDim.x + threadIdx.x; // over B*E*128
    if (t >= (long long)Bc * Ec * (HIDc / 4)) return;
    int chunk = (int)(t & 127);                       // 128 chunks of 4 features
    int eg    = (int)(t >> 7);                        // batched edge id
    int e = eg & (Ec - 1);
    int b = eg >> 16;
    int src = edge[e] + b * NBc;
    int dst = edge[Ec + e] + b * NBc;
    float nrm = dinv[src] * dinv[dst];
    const __bf16* hs = H1 + (size_t)src * HIDc + chunk * 4;
    float*        ao = AH + (size_t)dst * HIDc + chunk * 4;
    atomicAdd(&ao[0], nrm * (float)hs[0]);
    atomicAdd(&ao[1], nrm * (float)hs[1]);
    atomicAdd(&ao[2], nrm * (float)hs[2]);
    atomicAdd(&ao[3], nrm * (float)hs[3]);
}

__global__ void k_f32_to_bf16(const float* __restrict__ src, __bf16* __restrict__ dst, int n) {
    int t = blockIdx.x * blockDim.x + threadIdx.x;
    if (t < n) dst[t] = (__bf16)src[t];
}

__global__ void k_head(const __bf16* __restrict__ G, const float* __restrict__ Wh2,
                       const float* __restrict__ bh2, float* __restrict__ out) {
    int t = blockIdx.x * blockDim.x + threadIdx.x;   // over N*4
    if (t >= Nn * 4) return;
    int n = t >> 2;
    int c = t & 3;
    const __bf16* g = G + (size_t)n * HIDc;
    float s = bh2[c];
#pragma unroll 8
    for (int k = 0; k < HIDc; ++k)
        s += (float)g[k] * Wh2[k * 4 + c];
    if (c < 2) out[n * 2 + c] = tanhf(s);                  // log_s (S_MAX = 1.0)
    else       out[Nn * 2 + n * 2 + (c - 2)] = s;          // b
}

// ---------------------------------------------------------------------------
// WMMA fragment loaders (per CDNA5 ISA §7.12.2 16-bit A/B VGPR layouts)
// ---------------------------------------------------------------------------

__device__ __forceinline__ bf16x16 load_a_frag(const __bf16* sm, int stride, int rowBase, int lane) {
    const __bf16* row = sm + (size_t)(rowBase + (lane & 15)) * stride;
    int half = lane >> 4;
    union { bf16x16 v; unsigned u[8]; } f;
#pragma unroll
    for (int i = 0; i < 8; ++i) {
        int kk = (i < 4) ? (2 * i + 8 * half) : (16 + 2 * (i - 4) + 8 * half);
        f.u[i] = *(const unsigned*)(row + kk);
    }
    return f.v;
}

__device__ __forceinline__ bf16x16 load_b_frag(const __bf16* sm, int stride, int colBase, int lane) {
    const __bf16* row = sm + (size_t)(colBase + (lane & 15)) * stride;
    int half = lane >> 4;
    union { bf16x16 v; unsigned u[8]; } f;
#pragma unroll
    for (int i = 0; i < 8; ++i) {
        int kk = (i < 4) ? (2 * i + 8 * half) : (16 + 2 * (i - 4) + 8 * half);
        f.u[i] = *(const unsigned*)(row + kk);
    }
    return f.v;
}

// ---------------------------------------------------------------------------
// Fused (dual-input) WMMA GEMM:  C = silu(A0*W0 [+ A1*W1] + bias0 [+ bias1])
// A: (M x K) bf16 row-major.  W*t: (N x K) bf16 (pre-transposed, K-contiguous).
// Block = 256 threads = 8 waves; block tile 128(M) x 128(N); BK = 32.
// Each wave owns a 16(M) x 128(N) strip = 8 accumulators of v8f.
// ---------------------------------------------------------------------------
template <int DUAL>
__global__ void __launch_bounds__(256)
k_gemm_bias_silu(const __bf16* __restrict__ A0, const __bf16* __restrict__ W0t,
                 const float* __restrict__ bias0,
                 const __bf16* __restrict__ A1, const __bf16* __restrict__ W1t,
                 const float* __restrict__ bias1,
                 __bf16* __restrict__ C, int M, int N, int K) {
    constexpr int BM = 128, BN = 128, BK = 32;
    constexpr int STR = 40;                       // bf16 stride (80B row, uint4-aligned)

    __shared__ __align__(16) __bf16 sA0[BM * STR];
    __shared__ __align__(16) __bf16 sB0[BN * STR];
    __shared__ __align__(16) __bf16 sA1[DUAL ? BM * STR : 8];
    __shared__ __align__(16) __bf16 sB1[DUAL ? BN * STR : 8];

    const int tid    = threadIdx.x;
    const int lane   = tid & 31;
    const int wave   = tid >> 5;                  // 0..7
    const int blockM = blockIdx.x * BM;
    const int blockN = blockIdx.y * BN;

    floatx8 acc[8];
#pragma unroll
    for (int i = 0; i < 8; ++i)
#pragma unroll
        for (int j = 0; j < 8; ++j) acc[i][j] = 0.0f;

    for (int k0 = 0; k0 < K; k0 += BK) {
        // cooperative tile load: 128 rows x 32 bf16 per tile, uint4 chunks
#pragma unroll
        for (int i = tid; i < (BM * BK) / 8; i += 256) {
            int row = i >> 2;                     // BK/8 = 4 chunks per row
            int ch  = (i & 3) * 8;
            *(uint4*)(sA0 + row * STR + ch) =
                *(const uint4*)(A0 + (size_t)(blockM + row) * K + k0 + ch);
            *(uint4*)(sB0 + row * STR + ch) =
                *(const uint4*)(W0t + (size_t)(blockN + row) * K + k0 + ch);
            if constexpr (DUAL) {
                *(uint4*)(sA1 + row * STR + ch) =
                    *(const uint4*)(A1 + (size_t)(blockM + row) * K + k0 + ch);
                *(uint4*)(sB1 + row * STR + ch) =
                    *(const uint4*)(W1t + (size_t)(blockN + row) * K + k0 + ch);
            }
        }
        // prefetch next K tile into L2 while we compute this one
        if (k0 + BK < K) {
            int pr = tid >> 1;                    // 0..127 covers the tile rows
            __builtin_prefetch(A0 + (size_t)(blockM + pr) * K + k0 + BK, 0, 1);
            __builtin_prefetch(W0t + (size_t)(blockN + pr) * K + k0 + BK, 0, 1);
            if constexpr (DUAL) {
                __builtin_prefetch(A1 + (size_t)(blockM + pr) * K + k0 + BK, 0, 1);
                __builtin_prefetch(W1t + (size_t)(blockN + pr) * K + k0 + BK, 0, 1);
            }
        }
        __syncthreads();

        bf16x16 a0 = load_a_frag(sA0, STR, wave * 16, lane);
        bf16x16 a1{};
        if constexpr (DUAL) a1 = load_a_frag(sA1, STR, wave * 16, lane);

#pragma unroll
        for (int nt = 0; nt < 8; ++nt) {
            bf16x16 b0 = load_b_frag(sB0, STR, nt * 16, lane);
            acc[nt] = __builtin_amdgcn_wmma_f32_16x16x32_bf16(
                false, a0, false, b0, (short)0, acc[nt], false, false);
            if constexpr (DUAL) {
                bf16x16 b1 = load_b_frag(sB1, STR, nt * 16, lane);
                acc[nt] = __builtin_amdgcn_wmma_f32_16x16x32_bf16(
                    false, a1, false, b1, (short)0, acc[nt], false, false);
            }
        }
        __syncthreads();
    }

    // epilogue: bias + SiLU, store bf16 (C/D layout: lane%16 = N, vgpr v -> M)
    const int half = lane >> 4;
    const int colL = lane & 15;
#pragma unroll
    for (int nt = 0; nt < 8; ++nt) {
        int cn = blockN + nt * 16 + colL;
        float bsum = bias0[cn];
        if constexpr (DUAL) bsum += bias1[cn];
#pragma unroll
        for (int v = 0; v < 8; ++v) {
            int rm = blockM + wave * 16 + v + 8 * half;
            float x = acc[nt][v] + bsum;
            float s = x / (1.0f + __expf(-x));
            C[(size_t)rm * N + cn] = (__bf16)s;
        }
    }
}

// ---------------------------------------------------------------------------
// Host-side orchestration
// ---------------------------------------------------------------------------
extern "C" void kernel_launch(void* const* d_in, const int* in_sizes, int n_in,
                              void* d_out, int out_size, void* d_ws, size_t ws_size,
                              hipStream_t stream) {
    const float* X    = (const float*)d_in[0];
    const int*   edge = (const int*)d_in[1];
    const float* W1_0 = (const float*)d_in[2];
    const float* b1_0 = (const float*)d_in[3];
    const float* W1_1 = (const float*)d_in[4];
    const float* b1_1 = (const float*)d_in[5];
    const float* W2_0 = (const float*)d_in[6];
    const float* b2_0 = (const float*)d_in[7];
    const float* W2_1 = (const float*)d_in[8];
    const float* b2_1 = (const float*)d_in[9];
    const float* Wh1  = (const float*)d_in[10];
    const float* bh1  = (const float*)d_in[11];
    const float* Wh2  = (const float*)d_in[12];
    const float* bh2  = (const float*)d_in[13];
    float* out = (float*)d_out;

    // workspace carve-up (256B aligned)
    char*  ws = (char*)d_ws;
    size_t o  = 0;
    auto alloc = [&](size_t bytes) { size_t r = o; o = (o + bytes + 255) & ~(size_t)255; return r; };
    float*  dinv = (float*)(ws + alloc((size_t)Nn * 4));
    float*  ax   = (float*)(ws + alloc((size_t)Nn * 2 * 4));
    __bf16* H1b  = (__bf16*)(ws + alloc((size_t)Nn * HIDc * 2));
    float*  AHf  = (float*)(ws + alloc((size_t)Nn * HIDc * 4));   // reused as G
    __bf16* AHb  = (__bf16*)(ws + alloc((size_t)Nn * HIDc * 2));
    __bf16* H2b  = (__bf16*)(ws + alloc((size_t)Nn * HIDc * 2));
    __bf16* W20t = (__bf16*)(ws + alloc((size_t)HIDc * HIDc * 2));
    __bf16* W21t = (__bf16*)(ws + alloc((size_t)HIDc * HIDc * 2));
    __bf16* Wh1t = (__bf16*)(ws + alloc((size_t)HIDc * HIDc * 2));
    __bf16* Gb   = (__bf16*)AHf;                                   // alias (AHf dead by then)

    const int TB = 256;
    auto blocks = [&](long long n) { return (unsigned)((n + TB - 1) / TB); };

    // 1) normalization
    k_deg_init<<<blocks(Nn), TB, 0, stream>>>(dinv, Nn);
    k_deg_edges<<<blocks((long long)Bc * Ec), TB, 0, stream>>>(edge, dinv);
    k_dinv<<<blocks(Nn), TB, 0, stream>>>(dinv, Nn);

    // 2) layer-1 aggregation of the 2-wide input + dense 2->512 + SiLU
    k_ax_init<<<blocks((long long)Nn * 2), TB, 0, stream>>>(X, dinv, ax, Nn);
    k_ax_edges<<<blocks((long long)Bc * Ec), TB, 0, stream>>>(edge, X, dinv, ax);
    k_layer1<<<blocks((long long)Nn * HIDc), TB, 0, stream>>>(X, ax, W1_0, b1_0, W1_1, b1_1, H1b);

    // 3) weights -> bf16, N-major (K-contiguous) for WMMA B fragments
    k_wt_bf16<<<blocks((long long)HIDc * HIDc), TB, 0, stream>>>(W2_0, W20t, HIDc, HIDc);
    k_wt_bf16<<<blocks((long long)HIDc * HIDc), TB, 0, stream>>>(W2_1, W21t, HIDc, HIDc);
    k_wt_bf16<<<blocks((long long)HIDc * HIDc), TB, 0, stream>>>(Wh1, Wh1t, HIDc, HIDc);

    // 4) sparse aggregation of H1 (f32 atomics, destination lives in L2)
    k_agg_init<<<blocks((long long)Nn * HIDc), TB, 0, stream>>>(H1b, dinv, AHf);
    k_agg_edges<<<blocks((long long)Bc * Ec * (HIDc / 4)), TB, 0, stream>>>(edge, H1b, dinv, AHf);
    k_f32_to_bf16<<<blocks((long long)Nn * HIDc), TB, 0, stream>>>(AHf, AHb, Nn * HIDc);

    // 5) layer-2: dual-accumulate WMMA GEMM + bias + SiLU
    {
        dim3 grid(Nn / 128, HIDc / 128);
        k_gemm_bias_silu<1><<<grid, 256, 0, stream>>>(H1b, W20t, b2_0, AHb, W21t, b2_1,
                                                      H2b, Nn, HIDc, HIDc);
    }
    // 6) head hidden: single WMMA GEMM + bias + SiLU
    {
        dim3 grid(Nn / 128, HIDc / 128);
        k_gemm_bias_silu<0><<<grid, 256, 0, stream>>>(H2b, Wh1t, bh1, nullptr, nullptr, nullptr,
                                                      Gb, Nn, HIDc, HIDc);
    }
    // 7) head output 512->4, tanh split
    k_head<<<blocks((long long)Nn * 4), TB, 0, stream>>>(Gb, Wh2, bh2, out);
}
